// GNNMultiview_73718818669256
// MI455X (gfx1250) — matrix operand order repaired and checked by reading, not compiled, not run on hardware
//
#include <hip/hip_runtime.h>
#include <hip/hip_bf16.h>

// ---------------------------------------------------------------------------
// GNN multiview encoder for MI455X (gfx1250, wave32, WMMA).
// Conv layers 1..5 are GEMMs (M=co, K=512, N=positions) on
// v_wmma_f32_16x16x32_bf16; each wave computes 4 co-tiles per B fragment
// (4x B-operand reuse). GroupNorm stats come from in-kernel wave reductions
// (deterministic shfl tree) + a tiny second-stage reduce.
// ---------------------------------------------------------------------------

typedef __bf16  v16bf __attribute__((ext_vector_type(16)));
typedef float   v8f   __attribute__((ext_vector_type(8)));

#define NNODE 576           // B*CH = 64*9
#define CI    256

__device__ __forceinline__ float gelu_exact(float x) {
    return 0.5f * x * (1.0f + erff(x * 0.70710678118654752f));
}

// ---- pack conv weights [Co][256][2] f32 -> bf16 A-fragment layout ----------
// frag index p = ((co_tile*16 + kb32)*32 + lane)*16 + e
// lane: M = lane&15, kh = lane>>4 ; K = kb32*32 + kh*16 + e
// K order: k = tap*256 + ci  (matches B-fragment ci-contiguous loads)
__global__ void pack_w_k(const float* __restrict__ W,
                         __hip_bfloat16* __restrict__ packed, int Co) {
    int p = blockIdx.x * 256 + threadIdx.x;
    if (p >= Co * 512) return;
    int ct   = p >> 13;        // / 8192
    int r    = p & 8191;
    int kb32 = r >> 9;         // / 512
    int r2   = r & 511;
    int lane = r2 >> 4;
    int e    = r2 & 15;
    int M  = lane & 15, kh = lane >> 4;
    int K  = kb32 * 32 + kh * 16 + e;
    int co = ct * 16 + M;
    int tap = K >> 8, ci = K & 255;
    packed[p] = __float2bfloat16(W[(co * CI + ci) * 2 + tap]);
}

// ---- zero the two border rows of a padded bf16 activation buffer -----------
__global__ void zero_border_k(__hip_bfloat16* act, int Tpad, int C) {
    int i = blockIdx.x * 256 + threadIdx.x;
    int tot = NNODE * 2 * C;
    if (i >= tot) return;
    int c = i % C;
    int r = i / C;
    int n = r >> 1;
    int row = (r & 1) ? (Tpad - 1) : 0;
    act[((size_t)n * Tpad + row) * C + c] = __float2bfloat16(0.0f);
}

// ---- conv0: in_ch=1, w=3, stride 3, pad 1. Recompute trick (3 MACs/elem) ---
__global__ void conv0_stats_k(const float* __restrict__ x,
                              const float* __restrict__ w,
                              const float* __restrict__ b,
                              float* __restrict__ stats) {
    int n = blockIdx.x, co = threadIdx.x;
    const float w0 = w[co * 3], w1 = w[co * 3 + 1], w2 = w[co * 3 + 2], bb = b[co];
    const float* xr = x + (size_t)n * 3000;
    float s = 0.f, ss = 0.f;
    for (int t = 0; t < 1000; t++) {
        int p = 3 * t - 1;
        float v = bb + (p >= 0 ? w0 * xr[p] : 0.f) + w1 * xr[p + 1] + w2 * xr[p + 2];
        s += v; ss += v * v;
    }
    __shared__ float sh[256], sh2[256];
    sh[co] = s; sh2[co] = ss; __syncthreads();
    for (int o = 128; o; o >>= 1) {
        if (co < o) { sh[co] += sh[co + o]; sh2[co] += sh2[co + o]; }
        __syncthreads();
    }
    if (co == 0) { stats[2 * n] = sh[0]; stats[2 * n + 1] = sh2[0]; }
}

__global__ void conv0_gn_gelu_k(const float* __restrict__ x,
                                const float* __restrict__ w,
                                const float* __restrict__ b,
                                const float* __restrict__ g,
                                const float* __restrict__ be,
                                const float* __restrict__ stats,
                                __hip_bfloat16* __restrict__ act1) {
    int blk = blockIdx.x;
    int n = blk / 1000, t = blk % 1000, co = threadIdx.x;
    const float* xr = x + (size_t)n * 3000;
    int p = 3 * t - 1;
    float v = b[co] + (p >= 0 ? w[co * 3] * xr[p] : 0.f)
            + w[co * 3 + 1] * xr[p + 1] + w[co * 3 + 2] * xr[p + 2];
    float mu  = stats[2 * n] * (1.f / 256000.f);
    float var = stats[2 * n + 1] * (1.f / 256000.f) - mu * mu;
    float h = (v - mu) * rsqrtf(var + 1e-5f) * g[co] + be[co];
    act1[((size_t)n * 1002 + t + 1) * CI + co] = __float2bfloat16(gelu_exact(h));
}

// ---- WMMA conv: out[n][t][co] = sum_k W2[co,k] * act[n, 2t+tap, ci] + bias -
// One wave computes a 64co x 16t block (4 C tiles sharing each B fragment);
// K=512 -> 16 x 4 unrolled bf16 WMMAs. Also emits deterministic per-tile
// GroupNorm partial sums (fixed shfl_xor tree).
__global__ void __launch_bounds__(256)
conv_bf16_wmma_k(const __hip_bfloat16* __restrict__ act,     // [N][Tin+2][256]
                 const __hip_bfloat16* __restrict__ packedW, // fragment layout
                 const float* __restrict__ bias,
                 float* __restrict__ outBuf,                 // [N][Tout][Co] f32
                 float2* __restrict__ part,                  // [N][(Co/64)*nT]
                 int Tin, int Tout, int Co, int nT) {
    const int lane = threadIdx.x & 31, wave = threadIdx.x >> 5;
    const int n = blockIdx.z, co_blk = blockIdx.y;   // 64-co block
    const int t_tile = blockIdx.x * 8 + wave;
    if (t_tile >= nT) return;

    const int tN = lane & 15, kh = lane >> 4;
    const int t  = t_tile * 16 + tN;
    const int tc = t < Tout ? t : Tout - 1;   // clamp loads for masked lanes
    const bool valid = (t < Tout);

    const v16bf* pA = (const v16bf*)packedW + (size_t)co_blk * 2048 + lane;
    const __hip_bfloat16* actn = act + (size_t)n * (Tin + 2) * CI;

    v8f cacc[4];
    cacc[0] = {}; cacc[1] = {}; cacc[2] = {}; cacc[3] = {};
#pragma unroll
    for (int kb = 0; kb < 512; kb += 32) {
        const int kb32 = kb >> 5;
        const int tap  = kb >> 8;                              // 0 or 1
        const int ci0  = (kb & 255) + kh * 16;
        v16bf bf = *(const v16bf*)(actn + (size_t)(2 * tc + tap) * CI + ci0);
#pragma unroll
        for (int m = 0; m < 4; m++) {
            v16bf a = pA[m * 512 + kb32 * 32];                 // 2x b128 each
            cacc[m] = __builtin_amdgcn_wmma_f32_16x16x32_bf16(
                false, a, false, bf, (short)0, cacc[m], false, false);
        }
    }

    float s = 0.f, ss = 0.f;
    float* orow = outBuf + ((size_t)n * Tout + t) * Co;
#pragma unroll
    for (int m = 0; m < 4; m++) {
        const int co0 = (co_blk * 4 + m) * 16 + kh * 8;  // 8 contiguous co
        float4 b0 = *(const float4*)(bias + co0);
        float4 b1 = *(const float4*)(bias + co0 + 4);
        float r0 = cacc[m][0] + b0.x, r1 = cacc[m][1] + b0.y;
        float r2 = cacc[m][2] + b0.z, r3 = cacc[m][3] + b0.w;
        float r4 = cacc[m][4] + b1.x, r5 = cacc[m][5] + b1.y;
        float r6 = cacc[m][6] + b1.z, r7 = cacc[m][7] + b1.w;
        if (valid) {
            float4 o0 = { r0, r1, r2, r3 };
            float4 o1 = { r4, r5, r6, r7 };
            *(float4*)(orow + co0) = o0;
            *(float4*)(orow + co0 + 4) = o1;
            s  += r0 + r1 + r2 + r3 + r4 + r5 + r6 + r7;
            ss += r0*r0 + r1*r1 + r2*r2 + r3*r3 + r4*r4 + r5*r5 + r6*r6 + r7*r7;
        }
    }
    // deterministic wave reduction (fixed xor tree, identical every call)
#pragma unroll
    for (int msk = 16; msk; msk >>= 1) {
        s  += __shfl_xor(s, msk);
        ss += __shfl_xor(ss, msk);
    }
    if (lane == 0) {
        float2 v = { s, ss };
        part[((size_t)n * (Co >> 6) + co_blk) * nT + t_tile] = v;
    }
}

// ---- second-stage GroupNorm stats: reduce per-tile partials ----------------
__global__ void stats_reduce_k(const float2* __restrict__ part,
                               float* __restrict__ stats, int cnt) {
    int n = blockIdx.x, tid = threadIdx.x;
    const float2* p = part + (size_t)n * cnt;
    float s = 0.f, ss = 0.f;
    for (int i = tid; i < cnt; i += 256) { s += p[i].x; ss += p[i].y; }
    __shared__ float sh[256], sh2[256];
    sh[tid] = s; sh2[tid] = ss; __syncthreads();
    for (int o = 128; o; o >>= 1) {
        if (tid < o) { sh[tid] += sh[tid + o]; sh2[tid] += sh2[tid + o]; }
        __syncthreads();
    }
    if (!tid) { stats[2 * n] = sh[0]; stats[2 * n + 1] = sh2[0]; }
}

// ---- GN + GELU, pack to next layer's padded bf16 activation buffer ---------
__global__ void gn_gelu_pack_k(const float* __restrict__ in,
                               const float* __restrict__ stats,
                               const float* __restrict__ g,
                               const float* __restrict__ be,
                               __hip_bfloat16* __restrict__ actNext,
                               int Tout, int Co, float invCnt, int Tpad) {
    size_t idx = (size_t)blockIdx.x * 256 + threadIdx.x;
    size_t total = (size_t)NNODE * Tout * Co;
    if (idx >= total) return;
    int co = idx % Co;
    size_t r = idx / Co;
    int t = r % Tout;
    int n = r / Tout;
    float mu  = stats[2 * n] * invCnt;
    float var = stats[2 * n + 1] * invCnt - mu * mu;
    float h = (in[idx] - mu) * rsqrtf(var + 1e-5f) * g[co] + be[co];
    actNext[((size_t)n * Tpad + t + 1) * Co + co] = __float2bfloat16(gelu_exact(h));
}

// ---- GN + GELU, f32 output (after conv5, feeds 1x1 readout) ----------------
__global__ void gn_gelu_f32_k(const float* __restrict__ in,
                              const float* __restrict__ stats,
                              const float* __restrict__ g,
                              const float* __restrict__ be,
                              float* __restrict__ outp,
                              int Tout, int Co, float invCnt) {
    size_t idx = (size_t)blockIdx.x * 256 + threadIdx.x;
    size_t total = (size_t)NNODE * Tout * Co;
    if (idx >= total) return;
    int co = idx % Co;
    int n = (idx / Co) / Tout;
    float mu  = stats[2 * n] * invCnt;
    float var = stats[2 * n + 1] * invCnt - mu * mu;
    float h = (in[idx] - mu) * rsqrtf(var + 1e-5f) * g[co] + be[co];
    outp[idx] = gelu_exact(h);
}

// ---- 1x1 readout conv: lat[n,t,d] = sum_c a[n,t,c] * ro_w[c,d] + ro_b[d] ---
__global__ void readout_k(const float* __restrict__ a,
                          const float* __restrict__ w,
                          const float* __restrict__ b,
                          float* __restrict__ lat) {
    int idx = blockIdx.x * 256 + threadIdx.x;
    if (idx >= NNODE * 33 * 64) return;
    int d = idx & 63, row = idx >> 6;
    const float* ar = a + (size_t)row * 64;
    float acc = b[d];
#pragma unroll
    for (int c = 0; c < 64; c++) acc += ar[c] * w[c * 64 + d];
    lat[idx] = acc;
}

// ---- message precompute: A = lat @ W[:64], B = lat @ W[64:] ----------------
__global__ void msg_ab_k(const float* __restrict__ lat,
                         const float* __restrict__ W,
                         float* __restrict__ A, float* __restrict__ Bv) {
    int idx = blockIdx.x * 256 + threadIdx.x;
    if (idx >= NNODE * 33 * 64) return;
    int d = idx & 63, row = idx >> 6;
    const float* lr = lat + (size_t)row * 64;
    float a = 0.f, b = 0.f;
#pragma unroll
    for (int c = 0; c < 64; c++) {
        float v = lr[c];
        a += v * W[c * 64 + d];
        b += v * W[(64 + c) * 64 + d];
    }
    A[idx] = a; Bv[idx] = b;
}

// ---- message apply: lat'[k] = lat[k] + sum_{j!=k} relu(A_j + B_k + b)/8 ----
__global__ void msg_apply_k(const float* __restrict__ latIn,
                            const float* __restrict__ A,
                            const float* __restrict__ Bv,
                            const float* __restrict__ bias,
                            float* __restrict__ latOut) {
    int idx = blockIdx.x * 256 + threadIdx.x;
    if (idx >= NNODE * 33 * 64) return;
    int d = idx & 63, row = idx >> 6;
    int t = row % 33, n = row / 33;
    int b = n / 9, k = n % 9;
    float base = Bv[idx] + bias[d];
    float acc = 0.f;
#pragma unroll
    for (int j = 0; j < 9; j++) {
        if (j == k) continue;
        float m = A[(((size_t)(b * 9 + j)) * 33 + t) * 64 + d] + base;
        acc += fmaxf(m, 0.f);
    }
    latOut[idx] = latIn[idx] + acc * 0.125f;
}

// ---- head: segment-sum over 9 nodes, MLP, /9, projection, transpose --------
__global__ void head_k(const float* __restrict__ lat,
                       const float* __restrict__ w1, const float* __restrict__ b1,
                       const float* __restrict__ w2, const float* __restrict__ b2,
                       const float* __restrict__ pw, const float* __restrict__ pb,
                       float* __restrict__ out) {
    int bt = blockIdx.x;
    int b = bt / 33, t = bt % 33, d = threadIdx.x;
    __shared__ float y[64], y1[64], y2[64];
    float acc = 0.f;
    for (int k = 0; k < 9; k++)
        acc += lat[(((size_t)(b * 9 + k)) * 33 + t) * 64 + d];
    y[d] = acc; __syncthreads();
    float a1 = b1[d];
#pragma unroll
    for (int c = 0; c < 64; c++) a1 += y[c] * w1[c * 64 + d];
    y1[d] = fmaxf(a1, 0.f); __syncthreads();
    float a2 = b2[d];
#pragma unroll
    for (int c = 0; c < 64; c++) a2 += y1[c] * w2[c * 64 + d];
    y2[d] = a2 * (1.f / 9.f); __syncthreads();
    if (d < 32) {
        float o = pb[d];
#pragma unroll
        for (int c = 0; c < 64; c++) o += y2[c] * pw[c * 32 + d];
        out[((size_t)b * 32 + d) * 33 + t] = o;
    }
}

// ---------------------------------------------------------------------------
extern "C" void kernel_launch(void* const* d_in, const int* in_sizes, int n_in,
                              void* d_out, int out_size, void* d_ws, size_t ws_size,
                              hipStream_t stream) {
    (void)in_sizes; (void)n_in; (void)out_size; (void)ws_size;

    const float* x = (const float*)d_in[0];
    // conv_w{l}=1+4l  conv_b{l}=2+4l  gn_g{l}=3+4l  gn_b{l}=4+4l   (l=0..5)
    const float* ro_w = (const float*)d_in[25];
    const float* ro_b = (const float*)d_in[26];
    const float* msg_w0 = (const float*)d_in[27];
    const float* msg_b0 = (const float*)d_in[28];
    const float* msg_w1 = (const float*)d_in[29];
    const float* msg_b1 = (const float*)d_in[30];
    const float* rn_w1 = (const float*)d_in[31];
    const float* rn_b1 = (const float*)d_in[32];
    const float* rn_w2 = (const float*)d_in[33];
    const float* rn_b2 = (const float*)d_in[34];
    const float* proj_w = (const float*)d_in[35];
    const float* proj_b = (const float*)d_in[36];
    float* out = (float*)d_out;

    // layer geometry for conv layers 1..5 (all Ci=256, w=2, stride 2, pad 1)
    const int Tin_[5]  = { 1000, 501, 251, 126, 64 };
    const int Tout_[5] = {  501, 251, 126,  64, 33 };
    const int Co_[5]   = {  256, 256, 256, 256, 64 };

    // ---------------- workspace carve-up ----------------
    char* ws = (char*)d_ws;
    size_t off = 0;
    auto take = [&](size_t bytes) -> void* {
        void* p = ws + off;
        off = (off + bytes + 255) & ~(size_t)255;
        return p;
    };
    __hip_bfloat16* actb[5];
    for (int l = 0; l < 5; l++)
        actb[l] = (__hip_bfloat16*)take((size_t)NNODE * (Tin_[l] + 2) * CI * 2);
    float* convout = (float*)take((size_t)NNODE * 501 * 256 * 4);   // reused
    const size_t latBytes = (size_t)NNODE * 33 * 64 * 4;
    float* act6 = (float*)take(latBytes);
    float* lat0 = (float*)take(latBytes);
    float* lat1 = (float*)take(latBytes);
    float* Ab   = (float*)take(latBytes);
    float* Bb   = (float*)take(latBytes);
    __hip_bfloat16* pw[5];
    for (int l = 0; l < 5; l++)
        pw[l] = (__hip_bfloat16*)take((size_t)Co_[l] * 512 * 2);
    float* stats = (float*)take((size_t)6 * NNODE * 2 * 4);
    float2* part = (float2*)take((size_t)NNODE * 128 * sizeof(float2));

    // ---------------- per-call prologue ----------------
    for (int l = 0; l < 5; l++) {
        int tot = Co_[l] * 512;
        pack_w_k<<<(tot + 255) / 256, 256, 0, stream>>>(
            (const float*)d_in[1 + 4 * (l + 1)], pw[l], Co_[l]);
        int btot = NNODE * 2 * CI;
        zero_border_k<<<(btot + 255) / 256, 256, 0, stream>>>(
            actb[l], Tin_[l] + 2, CI);
    }

    // ---------------- conv0 (recompute trick) ----------------
    conv0_stats_k<<<NNODE, 256, 0, stream>>>(
        x, (const float*)d_in[1], (const float*)d_in[2], stats);
    conv0_gn_gelu_k<<<NNODE * 1000, 256, 0, stream>>>(
        x, (const float*)d_in[1], (const float*)d_in[2],
        (const float*)d_in[3], (const float*)d_in[4], stats, actb[0]);

    // ---------------- conv layers 1..5 (WMMA) ----------------
    for (int l = 0; l < 5; l++) {
        const int Tin = Tin_[l], Tout = Tout_[l], Co = Co_[l];
        float* statsL = stats + (size_t)(l + 1) * NNODE * 2;
        int nT = (Tout + 15) / 16;
        dim3 grid((nT + 7) / 8, Co / 64, NNODE);
        conv_bf16_wmma_k<<<grid, 256, 0, stream>>>(
            actb[l], pw[l], (const float*)d_in[2 + 4 * (l + 1)],
            convout, part, Tin, Tout, Co, nT);
        stats_reduce_k<<<NNODE, 256, 0, stream>>>(part, statsL, (Co >> 6) * nT);
        size_t total = (size_t)NNODE * Tout * Co;
        float invCnt = 1.0f / (float)(Tout * Co);
        if (l < 4) {
            gn_gelu_pack_k<<<(unsigned)((total + 255) / 256), 256, 0, stream>>>(
                convout, statsL, (const float*)d_in[3 + 4 * (l + 1)],
                (const float*)d_in[4 + 4 * (l + 1)], actb[l + 1],
                Tout, Co, invCnt, Tin_[l + 1] + 2);
        } else {
            gn_gelu_f32_k<<<(unsigned)((total + 255) / 256), 256, 0, stream>>>(
                convout, statsL, (const float*)d_in[3 + 4 * (l + 1)],
                (const float*)d_in[4 + 4 * (l + 1)], act6, Tout, Co, invCnt);
        }
    }

    // ---------------- readout + message passing + head ----------------
    int ntot = NNODE * 33 * 64;
    int nblk = (ntot + 255) / 256;
    readout_k<<<nblk, 256, 0, stream>>>(act6, ro_w, ro_b, lat0);

    msg_ab_k<<<nblk, 256, 0, stream>>>(lat0, msg_w0, Ab, Bb);
    msg_apply_k<<<nblk, 256, 0, stream>>>(lat0, Ab, Bb, msg_b0, lat1);

    msg_ab_k<<<nblk, 256, 0, stream>>>(lat1, msg_w1, Ab, Bb);
    msg_apply_k<<<nblk, 256, 0, stream>>>(lat1, Ab, Bb, msg_b1, lat0);

    head_k<<<64 * 33, 64, 0, stream>>>(lat0, rn_w1, rn_b1, rn_w2, rn_b2,
                                       proj_w, proj_b, out);
}
